// GCN_60636348285585
// MI455X (gfx1250) — compile-verified, matrix operand order checked by reading, not run on hardware
//
#include <hip/hip_runtime.h>
#include <hip/hip_bf16.h>

#define N_NODES 100000

typedef __attribute__((ext_vector_type(16))) __bf16 v16bf;
typedef __attribute__((ext_vector_type(8)))  float  v8f;

// ---------------------------------------------------------------------------
// Degree / normalization kernels
// ---------------------------------------------------------------------------
__global__ void deg_init(float* __restrict__ deg, int n) {
    int i = blockIdx.x * blockDim.x + threadIdx.x;
    if (i < n) deg[i] = 1.0f;  // self loop contributes 1 to every node
}

__global__ void deg_accum(const int* __restrict__ dst, float* deg, int E) {
    int i = blockIdx.x * blockDim.x + threadIdx.x;
    if (i < E) atomicAdd(&deg[dst[i]], 1.0f);
}

__global__ void deg_to_dinv(float* __restrict__ deg, int n) {
    int i = blockIdx.x * blockDim.x + threadIdx.x;
    if (i < n) {
        float d = deg[i];
        deg[i] = (d > 0.0f) ? rsqrtf(d) : 0.0f;   // in-place deg -> deg^{-1/2}
    }
}

// ---------------------------------------------------------------------------
// Split-bf16 WMMA GEMM:  C[nrows x NCOLS] = A[nrows x 128] * W[128 x NCOLS]
// One wave (32 threads) computes one 16x16 tile of C.
// A is decomposed as A_hi + A_lo (bf16 pair) so that
//   A*W ~= Ah*Wh + Ah*Wl + Al*Wh   (fp32 accumulate, ~2^-16 rel. error)
// VGPR packing per CDNA5 ISA 7.12.2:
//   A  : lane(0..15)=row M, khalf=lane>>4 selects K {kb..kb+7, kb+16..kb+23}
//   B  : lane(0..15)=col N, same K packing (column of W)
//   C/D: elem rr of v8f = row (khalf*8 + rr), col = lane&15
// ---------------------------------------------------------------------------
template <int NCOLS>
__global__ __launch_bounds__(32)
void gemm_wmma_bf16x3(const float* __restrict__ A,
                      const float* __restrict__ W,
                      float* __restrict__ C) {
    const int lane  = threadIdx.x & 31;
    const int sub   = lane & 15;
    const int khalf = lane >> 4;          // 0 or 1
    const int r     = blockIdx.x * 16 + sub;   // A row this lane feeds
    const int ncol  = blockIdx.y * 16 + sub;   // W column this lane feeds

    v8f c = {};
    #pragma unroll
    for (int kc = 0; kc < 128; kc += 32) {
        v16bf ah, al, bh, bl;
        #pragma unroll
        for (int j = 0; j < 8; ++j) {
            const int k0 = kc + khalf * 8 + j;        // K block 0
            const int k1 = k0 + 16;                   // K block 1
            // ---- A operand (row r) ----
            float a0 = A[r * 128 + k0];
            float a1 = A[r * 128 + k1];
            __bf16 a0h = (__bf16)a0, a1h = (__bf16)a1;
            ah[j]     = a0h;  ah[8 + j] = a1h;
            al[j]     = (__bf16)(a0 - (float)a0h);
            al[8 + j] = (__bf16)(a1 - (float)a1h);
            // ---- B operand (column ncol of W) ----
            float b0 = W[k0 * NCOLS + ncol];
            float b1 = W[k1 * NCOLS + ncol];
            __bf16 b0h = (__bf16)b0, b1h = (__bf16)b1;
            bh[j]     = b0h;  bh[8 + j] = b1h;
            bl[j]     = (__bf16)(b0 - (float)b0h);
            bl[8 + j] = (__bf16)(b1 - (float)b1h);
        }
        // D = A*B + C  (f32 accumulator), three split products
        c = __builtin_amdgcn_wmma_f32_16x16x32_bf16(false, ah, false, bh,
                                                    (short)0, c, false, false);
        c = __builtin_amdgcn_wmma_f32_16x16x32_bf16(false, ah, false, bl,
                                                    (short)0, c, false, false);
        c = __builtin_amdgcn_wmma_f32_16x16x32_bf16(false, al, false, bh,
                                                    (short)0, c, false, false);
    }

    #pragma unroll
    for (int rr = 0; rr < 8; ++rr) {
        const int m = blockIdx.x * 16 + khalf * 8 + rr;
        C[m * NCOLS + blockIdx.y * 16 + sub] = c[rr];
    }
}

// ---------------------------------------------------------------------------
// out[i,f] = h[i,f] * dinv[i]^2 + bias[f]      (self-loop term + bias;
// fully initializes the destination before edge atomics)
// ---------------------------------------------------------------------------
__global__ void prop_init(const float* __restrict__ h,
                          const float* __restrict__ dinv,
                          const float* __restrict__ bias,
                          float* __restrict__ out, int n, int ncols) {
    int i = blockIdx.x * blockDim.x + threadIdx.x;
    int total = n * ncols;
    if (i < total) {
        int node = i / ncols;
        int f    = i - node * ncols;
        float di = dinv[node];
        out[i] = h[i] * (di * di) + bias[f];
    }
}

// ---------------------------------------------------------------------------
// Edge scatter: out[dst] += h[src] * dinv[src]*dinv[dst]
// NCOLS/4 lanes per edge, each lane handles a float4 feature segment.
// ---------------------------------------------------------------------------
template <int NCOLS>
__global__ void prop_edges(const float* __restrict__ h,
                           const int* __restrict__ src,
                           const int* __restrict__ dst,
                           const float* __restrict__ dinv,
                           float* out, int E) {
    const int LPE = NCOLS / 4;
    long long t = (long long)blockIdx.x * blockDim.x + threadIdx.x;
    long long total = (long long)E * LPE;
    if (t >= total) return;
    int e   = (int)(t / LPE);
    int seg = (int)(t - (long long)e * LPE);
    int s = src[e];
    int d = dst[e];
    float norm = dinv[s] * dinv[d];
    const float4 hv = *(const float4*)(h + (long long)s * NCOLS + seg * 4);
    float* o = out + (long long)d * NCOLS + seg * 4;
    atomicAdd(o + 0, hv.x * norm);
    atomicAdd(o + 1, hv.y * norm);
    atomicAdd(o + 2, hv.z * norm);
    atomicAdd(o + 3, hv.w * norm);
}

// ---------------------------------------------------------------------------
// Orchestration
// ---------------------------------------------------------------------------
extern "C" void kernel_launch(void* const* d_in, const int* in_sizes, int n_in,
                              void* d_out, int out_size, void* d_ws, size_t ws_size,
                              hipStream_t stream) {
    const float* x  = (const float*)d_in[0];
    const int*   ei = (const int*)d_in[1];
    const float* W1 = (const float*)d_in[2];
    const float* b1 = (const float*)d_in[3];
    const float* W2 = (const float*)d_in[4];
    const float* b2 = (const float*)d_in[5];
    float* out = (float*)d_out;

    const int N = N_NODES;
    const int E = in_sizes[1] / 2;       // edge_index is [2, E]
    const int* src = ei;
    const int* dst = ei + E;

    // workspace layout (floats): dinv | h1[N*128] | p1[N*128] | h2[N*64]
    float* dinv = (float*)d_ws;
    float* h1 = dinv + ((N + 255) & ~255);
    float* p1 = h1 + (size_t)N * 128;
    float* h2 = p1 + (size_t)N * 128;

    // ---- normalization: deg -> deg^{-1/2} ----
    deg_init   <<<(N + 255) / 256, 256, 0, stream>>>(dinv, N);
    deg_accum  <<<(E + 255) / 256, 256, 0, stream>>>(dst, dinv, E);
    deg_to_dinv<<<(N + 255) / 256, 256, 0, stream>>>(dinv, N);

    // ---- layer 1: h1 = x @ W1 ; p1 = propagate(h1) + b1 ----
    gemm_wmma_bf16x3<128><<<dim3(N / 16, 128 / 16), 32, 0, stream>>>(x, W1, h1);
    prop_init<<<((N * 128) + 255) / 256, 256, 0, stream>>>(h1, dinv, b1, p1, N, 128);
    {
        long long total = (long long)E * 32;
        prop_edges<128><<<(unsigned)((total + 255) / 256), 256, 0, stream>>>(
            h1, src, dst, dinv, p1, E);
    }

    // ---- layer 2: h2 = p1 @ W2 ; out = propagate(h2) + b2 ----
    gemm_wmma_bf16x3<64><<<dim3(N / 16, 64 / 16), 32, 0, stream>>>(p1, W2, h2);
    prop_init<<<((N * 64) + 255) / 256, 256, 0, stream>>>(h2, dinv, b2, out, N, 64);
    {
        long long total = (long long)E * 16;
        prop_edges<64><<<(unsigned)((total + 255) / 256), 256, 0, stream>>>(
            h2, src, dst, dinv, out, E);
    }
}
